// MultiHeadAttentionWithRoPE_56676388438113
// MI455X (gfx1250) — compile-verified, matrix operand order checked by reading, not run on hardware
//
#include <hip/hip_runtime.h>
#include <hip/hip_bf16.h>

#define T_SEQ 1024
#define BATCH 4
#define NHEAD 16
#define HDIM  64
#define EMB   1024

typedef __attribute__((ext_vector_type(16))) __bf16 v16bf;
typedef __attribute__((ext_vector_type(8)))  __bf16 v8bf;
typedef __attribute__((ext_vector_type(8)))  float  v8f;

#define WMMA_BF16(a, b, c) \
  __builtin_amdgcn_wmma_f32_16x16x32_bf16(false, (a), false, (b), (short)0, (c), false, false)

// A/B fragment (16x32 bf16) from a bf16 row pointer; per-lane K chunks at k0 and k0+16.
static __device__ __forceinline__ v16bf frag_bf16(const __bf16* row, int k0) {
  union { v16bf v; uint4 u[2]; } f;
  f.u[0] = *(const uint4*)(row + k0);
  f.u[1] = *(const uint4*)(row + k0 + 16);
  return f.v;
}

static __device__ __forceinline__ float red_max16(float v) {
  v = fmaxf(v, __shfl_xor(v, 1, 32));
  v = fmaxf(v, __shfl_xor(v, 2, 32));
  v = fmaxf(v, __shfl_xor(v, 4, 32));
  v = fmaxf(v, __shfl_xor(v, 8, 32));
  return v;
}
static __device__ __forceinline__ float red_sum16(float v) {
  v += __shfl_xor(v, 1, 32);
  v += __shfl_xor(v, 2, 32);
  v += __shfl_xor(v, 4, 32);
  v += __shfl_xor(v, 8, 32);
  return v;
}

// ---------------------------------------------------------------------------
// Kernel 0: one-shot f32 -> bf16 conversion (8 elements / thread, b128 I/O).
// ---------------------------------------------------------------------------
__global__ __launch_bounds__(256) void cvt_f32_bf16(const float* __restrict__ s,
                                                    __bf16* __restrict__ d, int n8) {
  const int i = blockIdx.x * 256 + threadIdx.x;
  if (i >= n8) return;
  float4 a = ((const float4*)s)[2 * i];
  float4 b = ((const float4*)s)[2 * i + 1];
  union { v8bf v; uint4 u; } o;
  o.v[0] = (__bf16)a.x; o.v[1] = (__bf16)a.y; o.v[2] = (__bf16)a.z; o.v[3] = (__bf16)a.w;
  o.v[4] = (__bf16)b.x; o.v[5] = (__bf16)b.y; o.v[6] = (__bf16)b.z; o.v[7] = (__bf16)b.w;
  ((uint4*)d)[i] = o.u;
}

// ---------------------------------------------------------------------------
// Kernel 1: fused QKV projection (+RoPE for Q,K). blockIdx.y = 0(Q)/1(K)/2(V).
// Per wave: 32x64 output tile (2 A-frags x 4 B-frags, 8 WMMAs per 32-K step).
// Output layouts: Q,K -> bf16 [b][h][t][d] ; V -> bf16 [b][h][d][t] (transposed)
// ---------------------------------------------------------------------------
__global__ __launch_bounds__(256) void qkv_proj_rope(
    const __bf16* __restrict__ Xq, const __bf16* __restrict__ Xk, const __bf16* __restrict__ Xv,
    const __bf16* __restrict__ Wq, const __bf16* __restrict__ Wk, const __bf16* __restrict__ Wv,
    __bf16* __restrict__ Qb, __bf16* __restrict__ Kb, __bf16* __restrict__ Vt) {
  const int proj = blockIdx.y;
  const __bf16* X = proj == 0 ? Xq : (proj == 1 ? Xk : Xv);
  const __bf16* W = proj == 0 ? Wq : (proj == 1 ? Wk : Wv);

  const int lane = threadIdx.x & 31;
  const int wave = threadIdx.x >> 5;
  const int lm   = lane & 15;
  const int hi   = lane >> 4;
  const int pat  = hi ? 8 : 0;

  const int tile  = blockIdx.x * 8 + wave;  // 2048 wave-tiles
  const int mtile = tile >> 4;              // 128 row tiles of 32
  const int h     = tile & 15;              // head == 64-wide column tile
  const int mbase = mtile * 32;
  const int cbase = h * 64;

  const __bf16* ar0 = X + (size_t)(mbase + lm) * EMB;
  const __bf16* ar1 = X + (size_t)(mbase + 16 + lm) * EMB;
  const __bf16* wr0 = W + (size_t)(cbase +  0 + lm) * EMB;
  const __bf16* wr1 = W + (size_t)(cbase + 16 + lm) * EMB;
  const __bf16* wr2 = W + (size_t)(cbase + 32 + lm) * EMB;
  const __bf16* wr3 = W + (size_t)(cbase + 48 + lm) * EMB;

  v8f acc[2][4];
#pragma unroll
  for (int mi = 0; mi < 2; ++mi)
#pragma unroll
    for (int s = 0; s < 4; ++s) acc[mi][s] = (v8f){0,0,0,0,0,0,0,0};

  for (int kk = 0; kk < EMB; kk += 32) {
    v16bf a0 = frag_bf16(ar0, kk + pat);
    v16bf a1 = frag_bf16(ar1, kk + pat);
    v16bf b0 = frag_bf16(wr0, kk + pat);
    v16bf b1 = frag_bf16(wr1, kk + pat);
    v16bf b2 = frag_bf16(wr2, kk + pat);
    v16bf b3 = frag_bf16(wr3, kk + pat);
    acc[0][0] = WMMA_BF16(a0, b0, acc[0][0]);
    acc[0][1] = WMMA_BF16(a0, b1, acc[0][1]);
    acc[0][2] = WMMA_BF16(a0, b2, acc[0][2]);
    acc[0][3] = WMMA_BF16(a0, b3, acc[0][3]);
    acc[1][0] = WMMA_BF16(a1, b0, acc[1][0]);
    acc[1][1] = WMMA_BF16(a1, b1, acc[1][1]);
    acc[1][2] = WMMA_BF16(a1, b2, acc[1][2]);
    acc[1][3] = WMMA_BF16(a1, b3, acc[1][3]);
  }

#pragma unroll
  for (int mi = 0; mi < 2; ++mi) {
    if (proj < 2) {
      __bf16* Dst = proj == 0 ? Qb : Kb;
#pragma unroll
      for (int r = 0; r < 8; ++r) {
        const int rm = mbase + mi * 16 + r + hi * 8;
        const int t  = rm >> 2;   // rows of X are t*B + b
        const int b  = rm & 3;
        __bf16* orow = Dst + (((size_t)(b * NHEAD + h)) * T_SEQ + t) * HDIM;
        // RoPE pairs (d, d+32): acc[][0]<->acc[][2] holds d=lm / lm+32; [1]<->[3]: lm+16 / lm+48
        float sn, cs;
        float th = __expf((float)lm * -0.28782313662425575f);  // 10000^(-d/32)
        __sincosf((float)t * th, &sn, &cs);
        float x1 = acc[mi][0][r], x2 = acc[mi][2][r];
        orow[lm]      = (__bf16)(x1 * cs - x2 * sn);
        orow[lm + 32] = (__bf16)(x1 * sn + x2 * cs);
        th = __expf((float)(lm + 16) * -0.28782313662425575f);
        __sincosf((float)t * th, &sn, &cs);
        x1 = acc[mi][1][r]; x2 = acc[mi][3][r];
        orow[lm + 16] = (__bf16)(x1 * cs - x2 * sn);
        orow[lm + 48] = (__bf16)(x1 * sn + x2 * cs);
      }
    } else {
#pragma unroll
      for (int r = 0; r < 8; ++r) {
        const int rm = mbase + mi * 16 + r + hi * 8;
        const int t  = rm >> 2;
        const int b  = rm & 3;
        __bf16* vcol = Vt + ((size_t)(b * NHEAD + h)) * HDIM * T_SEQ;
        vcol[(size_t)(lm +  0) * T_SEQ + t] = (__bf16)acc[mi][0][r];
        vcol[(size_t)(lm + 16) * T_SEQ + t] = (__bf16)acc[mi][1][r];
        vcol[(size_t)(lm + 32) * T_SEQ + t] = (__bf16)acc[mi][2][r];
        vcol[(size_t)(lm + 48) * T_SEQ + t] = (__bf16)acc[mi][3][r];
      }
    }
  }
}

// ---------------------------------------------------------------------------
// Kernel 2: flash attention. One wave = 16 query rows of one (b,h).
// Output: bf16 [t][b][h*64+d]  (row order matches the final GEMM's A matrix)
// ---------------------------------------------------------------------------
__global__ __launch_bounds__(256) void flash_attn(
    const __bf16* __restrict__ Qb, const __bf16* __restrict__ Kb,
    const __bf16* __restrict__ Vt,
    const int* __restrict__ kpm, const int* __restrict__ amask,
    __bf16* __restrict__ Ob) {
  __shared__ __align__(16) __bf16 Psh[8][16][32];

  const int lane = threadIdx.x & 31;
  const int wave = threadIdx.x >> 5;
  const int lm   = lane & 15;
  const int hi   = lane >> 4;
  const int pat  = hi ? 8 : 0;

  const int tile  = blockIdx.x * 8 + wave;  // 4096 = B*H*(T/16)
  const int qt    = tile & 63;
  const int bh    = tile >> 6;
  const int b     = bh >> 4;
  const int h     = bh & 15;
  const int qbase = qt * 16;

  const __bf16* Qh = Qb + (size_t)bh * T_SEQ * HDIM;
  const __bf16* Kh = Kb + (size_t)bh * T_SEQ * HDIM;
  const __bf16* Vh = Vt + (size_t)bh * HDIM * T_SEQ;
  const int* amb = amask + (size_t)b * T_SEQ * T_SEQ;
  const int* kpb = kpm + (size_t)b * T_SEQ;

  const __bf16* qrow = Qh + (size_t)(qbase + lm) * HDIM;
  const v16bf qa0 = frag_bf16(qrow, pat);
  const v16bf qa1 = frag_bf16(qrow, 32 + pat);

  v8f o0 = {0,0,0,0,0,0,0,0};
  v8f o1 = o0, o2 = o0, o3 = o0;
  float rmax[8], rsum[8];
#pragma unroll
  for (int r = 0; r < 8; ++r) { rmax[r] = -1e30f; rsum[r] = 0.0f; }

  const float scale = 0.125f;       // HEAD_DIM^-0.5
  const float NEGM  = -100000.0f;

  for (int n0 = 0; n0 < T_SEQ; n0 += 32) {
    // ---- scores for two 16-key subtiles (K B-fragments are rows of K) ----
    const __bf16* krA = Kh + (size_t)(n0 + lm) * HDIM;
    const __bf16* krB = Kh + (size_t)(n0 + 16 + lm) * HDIM;
    // prefetch next chunk's K/V rows (lowers to global_prefetch_b8)
    __builtin_prefetch(krA + 32 * HDIM, 0, 1);
    __builtin_prefetch(krB + 32 * HDIM, 0, 1);
    __builtin_prefetch(Vh + (size_t)lm * T_SEQ + n0 + 32, 0, 1);

    v8f z = {0,0,0,0,0,0,0,0};
    v8f sa = WMMA_BF16(qa0, frag_bf16(krA, pat), z);
    sa     = WMMA_BF16(qa1, frag_bf16(krA, 32 + pat), sa);
    v8f sb = WMMA_BF16(qa0, frag_bf16(krB, pat), z);
    sb     = WMMA_BF16(qa1, frag_bf16(krB, 32 + pat), sb);

    const float mkA = (float)kpb[n0 + lm] * NEGM;
    const float mkB = (float)kpb[n0 + 16 + lm] * NEGM;

    // ---- online softmax per row; stage P tile into LDS in A-layout ----
#pragma unroll
    for (int r = 0; r < 8; ++r) {
      const int m = r + hi * 8;
      const int q = qbase + m;
      const int* am = amb + (size_t)q * T_SEQ + n0;
      float va = sa[r] * scale + (float)am[lm] * NEGM + mkA;
      float vb = sb[r] * scale + (float)am[lm + 16] * NEGM + mkB;
      float tm   = red_max16(fmaxf(va, vb));
      float nm   = fmaxf(rmax[r], tm);
      float corr = __expf(rmax[r] - nm);
      rmax[r] = nm;
      float pa = __expf(va - nm);
      float pb = __expf(vb - nm);
      rsum[r] = rsum[r] * corr + red_sum16(pa + pb);
      o0[r] *= corr; o1[r] *= corr; o2[r] *= corr; o3[r] *= corr;
      Psh[wave][m][lm]      = (__bf16)pa;
      Psh[wave][m][lm + 16] = (__bf16)pb;
    }
    // intra-wave LDS store -> load ordering (cross-lane dep invisible to compiler)
    asm volatile("s_wait_dscnt 0x0" ::: "memory");

    // ---- P (16x32) @ V (32x64): V B-fragments are rows of transposed V ----
    const v16bf pfr = frag_bf16(&Psh[wave][lm][0], pat);
    o0 = WMMA_BF16(pfr, frag_bf16(Vh + (size_t)(lm +  0) * T_SEQ, n0 + pat), o0);
    o1 = WMMA_BF16(pfr, frag_bf16(Vh + (size_t)(lm + 16) * T_SEQ, n0 + pat), o1);
    o2 = WMMA_BF16(pfr, frag_bf16(Vh + (size_t)(lm + 32) * T_SEQ, n0 + pat), o2);
    o3 = WMMA_BF16(pfr, frag_bf16(Vh + (size_t)(lm + 48) * T_SEQ, n0 + pat), o3);
    asm volatile("" ::: "memory");
  }

  // ---- normalize and store in (t, b, e) row order for the output GEMM ----
#pragma unroll
  for (int r = 0; r < 8; ++r) {
    const int q = qbase + r + hi * 8;
    const float inv = 1.0f / rsum[r];
    __bf16* od = Ob + ((size_t)q * BATCH + b) * EMB + h * HDIM;
    od[lm]      = (__bf16)(o0[r] * inv);
    od[lm + 16] = (__bf16)(o1[r] * inv);
    od[lm + 32] = (__bf16)(o2[r] * inv);
    od[lm + 48] = (__bf16)(o3[r] * inv);
  }
}

// ---------------------------------------------------------------------------
// Kernel 3: output projection  out = Ob(4096x1024 bf16) @ o_w^T  -> f32
// Per wave: 32x64 output tile.
// ---------------------------------------------------------------------------
__global__ __launch_bounds__(256) void out_proj(
    const __bf16* __restrict__ Ob, const __bf16* __restrict__ Ow,
    float* __restrict__ out) {
  const int lane = threadIdx.x & 31;
  const int wave = threadIdx.x >> 5;
  const int lm   = lane & 15;
  const int hi   = lane >> 4;
  const int pat  = hi ? 8 : 0;

  const int tile  = blockIdx.x * 8 + wave;  // 2048
  const int mtile = tile >> 4;
  const int ntile = tile & 15;
  const int mbase = mtile * 32;
  const int cbase = ntile * 64;

  const __bf16* ar0 = Ob + (size_t)(mbase + lm) * EMB;
  const __bf16* ar1 = Ob + (size_t)(mbase + 16 + lm) * EMB;
  const __bf16* wr0 = Ow + (size_t)(cbase +  0 + lm) * EMB;
  const __bf16* wr1 = Ow + (size_t)(cbase + 16 + lm) * EMB;
  const __bf16* wr2 = Ow + (size_t)(cbase + 32 + lm) * EMB;
  const __bf16* wr3 = Ow + (size_t)(cbase + 48 + lm) * EMB;

  v8f acc[2][4];
#pragma unroll
  for (int mi = 0; mi < 2; ++mi)
#pragma unroll
    for (int s = 0; s < 4; ++s) acc[mi][s] = (v8f){0,0,0,0,0,0,0,0};

  for (int kk = 0; kk < EMB; kk += 32) {
    v16bf a0 = frag_bf16(ar0, kk + pat);
    v16bf a1 = frag_bf16(ar1, kk + pat);
    v16bf b0 = frag_bf16(wr0, kk + pat);
    v16bf b1 = frag_bf16(wr1, kk + pat);
    v16bf b2 = frag_bf16(wr2, kk + pat);
    v16bf b3 = frag_bf16(wr3, kk + pat);
    acc[0][0] = WMMA_BF16(a0, b0, acc[0][0]);
    acc[0][1] = WMMA_BF16(a0, b1, acc[0][1]);
    acc[0][2] = WMMA_BF16(a0, b2, acc[0][2]);
    acc[0][3] = WMMA_BF16(a0, b3, acc[0][3]);
    acc[1][0] = WMMA_BF16(a1, b0, acc[1][0]);
    acc[1][1] = WMMA_BF16(a1, b1, acc[1][1]);
    acc[1][2] = WMMA_BF16(a1, b2, acc[1][2]);
    acc[1][3] = WMMA_BF16(a1, b3, acc[1][3]);
  }

#pragma unroll
  for (int mi = 0; mi < 2; ++mi) {
#pragma unroll
    for (int r = 0; r < 8; ++r) {
      const int rm = mbase + mi * 16 + r + hi * 8;
      float* orow = out + (size_t)rm * EMB + cbase;
      orow[lm]      = acc[mi][0][r];
      orow[lm + 16] = acc[mi][1][r];
      orow[lm + 32] = acc[mi][2][r];
      orow[lm + 48] = acc[mi][3][r];
    }
  }
}

// ---------------------------------------------------------------------------
extern "C" void kernel_launch(void* const* d_in, const int* in_sizes, int n_in,
                              void* d_out, int out_size, void* d_ws, size_t ws_size,
                              hipStream_t stream) {
  const float* q   = (const float*)d_in[0];
  const float* k   = (const float*)d_in[1];
  const float* v   = (const float*)d_in[2];
  const int*   kpm = (const int*)d_in[3];
  const int*   am  = (const int*)d_in[4];
  const float* qw  = (const float*)d_in[5];
  const float* kw  = (const float*)d_in[6];
  const float* vw  = (const float*)d_in[7];
  const float* ow  = (const float*)d_in[8];
  float* out = (float*)d_out;

  const size_t MB = 1024u * 1024u;
  char* ws = (char*)d_ws;
  __bf16* Xq = (__bf16*)(ws);            // 8 MB  (t,b,e) bf16   [aliased by Ob later]
  __bf16* Xk = (__bf16*)(ws +  8 * MB);  // 8 MB
  __bf16* Xv = (__bf16*)(ws + 16 * MB);  // 8 MB
  __bf16* Wq = (__bf16*)(ws + 24 * MB);  // 2 MB  (n,k) bf16
  __bf16* Wk = (__bf16*)(ws + 26 * MB);  // 2 MB
  __bf16* Wv = (__bf16*)(ws + 28 * MB);  // 2 MB
  __bf16* Wo = (__bf16*)(ws + 30 * MB);  // 2 MB
  __bf16* Qb = (__bf16*)(ws + 32 * MB);  // 8 MB  [b][h][t][d]
  __bf16* Kb = (__bf16*)(ws + 40 * MB);  // 8 MB  [b][h][t][d]
  __bf16* Vt = (__bf16*)(ws + 48 * MB);  // 8 MB  [b][h][d][t]
  __bf16* Ob = (__bf16*)(ws);            // 8 MB  [t][b][e]   (Xq is dead by then)

  const int n8x = T_SEQ * BATCH * EMB / 8;  // 524288
  const int n8w = EMB * EMB / 8;            // 131072
  cvt_f32_bf16<<<n8x / 256, 256, 0, stream>>>(q, Xq, n8x);
  cvt_f32_bf16<<<n8x / 256, 256, 0, stream>>>(k, Xk, n8x);
  cvt_f32_bf16<<<n8x / 256, 256, 0, stream>>>(v, Xv, n8x);
  cvt_f32_bf16<<<n8w / 256, 256, 0, stream>>>(qw, Wq, n8w);
  cvt_f32_bf16<<<n8w / 256, 256, 0, stream>>>(kw, Wk, n8w);
  cvt_f32_bf16<<<n8w / 256, 256, 0, stream>>>(vw, Wv, n8w);
  cvt_f32_bf16<<<n8w / 256, 256, 0, stream>>>(ow, Wo, n8w);

  dim3 g1(256, 3, 1);
  qkv_proj_rope<<<g1, 256, 0, stream>>>(Xq, Xk, Xv, Wq, Wk, Wv, Qb, Kb, Vt);
  flash_attn<<<512, 256, 0, stream>>>(Qb, Kb, Vt, kpm, am, Ob);
  out_proj<<<256, 256, 0, stream>>>(Ob, Wo, out);
}
